// MultiheadAttention_39092792328758
// MI455X (gfx1250) — matrix operand order repaired and checked by reading, not compiled, not run on hardware
//
#include <hip/hip_runtime.h>

#define Bn 2
#define Tn 2048
#define Dn 1024
#define Hn 16
#define DHn 64
#define BTn (Bn * Tn)

typedef __attribute__((ext_vector_type(16))) __bf16 v16bf;
typedef __attribute__((ext_vector_type(8)))  __bf16 v8bf;
typedef __attribute__((ext_vector_type(4)))  __bf16 v4bf;
typedef __attribute__((ext_vector_type(8)))  float  v8f;

union V16U { v16bf v; v8bf h[2]; };

// A-matrix 16x32 bf16 fragment (row-major source, stride in elements).
// lane l: row = l&15; K chunks at elem offsets (l>>4)*8 and 16+(l>>4)*8.
__device__ __forceinline__ v16bf load_a_frag(const __bf16* base, int stride, int lane) {
  const int row = lane & 15, half = lane >> 4;
  const __bf16* p = base + row * stride + half * 8;
  V16U u;
  u.h[0] = *(const v8bf*)(p);
  u.h[1] = *(const v8bf*)(p + 16);
  return u.v;
}

// B-matrix 32x16 bf16 fragment. Memory holds B^T row-major (row n = column n of B,
// K contiguous). lane l: n = l&15; K chunks at elem offsets (l>>4)*8 and 16+(l>>4)*8.
__device__ __forceinline__ v16bf load_b_frag(const __bf16* base, int stride, int lane) {
  const __bf16* p = base + (lane & 15) * stride + (lane >> 4) * 8;
  V16U u;
  u.h[0] = *(const v8bf*)(p);
  u.h[1] = *(const v8bf*)(p + 16);
  return u.v;
}

__device__ __forceinline__ v8f wmma_bf16(v16bf a, v16bf b, v8f c) {
  return __builtin_amdgcn_wmma_f32_16x16x32_bf16(false, a, false, b, (short)0, c,
                                                 false, false);
}

// ---------------------------------------------------------------- f32 -> bf16
__global__ void cvt_f32_bf16(const float* __restrict__ in, __bf16* __restrict__ out,
                             int n) {
  const int i = (blockIdx.x * blockDim.x + threadIdx.x) * 4;
  if (i + 3 < n) {
    const float4 v = *(const float4*)(in + i);
    v4bf o;
    o[0] = (__bf16)v.x; o[1] = (__bf16)v.y; o[2] = (__bf16)v.z; o[3] = (__bf16)v.w;
    *(v4bf*)(out + i) = o;
  }
}

// ------------------------------------------------- Q/K/V projection (x @ W.T)
// grid: (BT/256, H, 3) block 256 (8 waves). Wave computes 32 rows x 64 cols.
// k-loop unrolled x2 with ping-pong buffers; each buffer is reloaded right
// after its WMMAs consume it -> loads land in-place (no swap copies) and have
// a full half-iteration (8 WMMAs) to cover their latency.
// mode 0 -> Q [B,H,T,DH], mode 1 -> K [B,H,T,DH], mode 2 -> V^T [B,H,DH,T].
__global__ __launch_bounds__(256) void qkv_proj_kernel(
    const __bf16* __restrict__ X, const __bf16* __restrict__ Wq,
    const __bf16* __restrict__ Wk, const __bf16* __restrict__ Wv,
    __bf16* __restrict__ Q, __bf16* __restrict__ K, __bf16* __restrict__ V) {
  const int lane = threadIdx.x & 31;
  const int w    = threadIdx.x >> 5;
  const int m0   = blockIdx.x * 256 + w * 32;
  const int h    = blockIdx.y;
  const int n0   = h * 64;
  const int mode = blockIdx.z;
  const __bf16* W = (mode == 0) ? Wq : (mode == 1) ? Wk : Wv;

  const __bf16* Ar0 = X + (size_t)m0 * Dn;
  const __bf16* Ar1 = Ar0 + (size_t)16 * Dn;

  v8f acc[2][4] = {};
  v16bf a0[2], a1[2], b0[4], b1[4];
  a0[0] = load_a_frag(Ar0, Dn, lane);
  a0[1] = load_a_frag(Ar1, Dn, lane);
  a1[0] = load_a_frag(Ar0 + 32, Dn, lane);
  a1[1] = load_a_frag(Ar1 + 32, Dn, lane);
#pragma unroll
  for (int i = 0; i < 4; ++i) {
    b0[i] = load_b_frag(W + (size_t)(n0 + i * 16) * Dn, Dn, lane);
    b1[i] = load_b_frag(W + (size_t)(n0 + i * 16) * Dn + 32, Dn, lane);
  }

  for (int k = 0; k < Dn; k += 64) {
    const int kn0 = (k + 64 < Dn) ? k + 64 : 0;  // clamp: redundant in-bounds load
    const int kn1 = (k + 96 < Dn) ? k + 96 : 0;
    // ---- half 0: consume buf0 (depth k), then refill it for depth k+64
#pragma unroll
    for (int i = 0; i < 4; ++i) acc[0][i] = wmma_bf16(a0[0], b0[i], acc[0][i]);
#pragma unroll
    for (int i = 0; i < 4; ++i) acc[1][i] = wmma_bf16(a0[1], b0[i], acc[1][i]);
    a0[0] = load_a_frag(Ar0 + kn0, Dn, lane);
    a0[1] = load_a_frag(Ar1 + kn0, Dn, lane);
#pragma unroll
    for (int i = 0; i < 4; ++i)
      b0[i] = load_b_frag(W + (size_t)(n0 + i * 16) * Dn + kn0, Dn, lane);
    // ---- half 1: consume buf1 (depth k+32), then refill it for depth k+96
#pragma unroll
    for (int i = 0; i < 4; ++i) acc[0][i] = wmma_bf16(a1[0], b1[i], acc[0][i]);
#pragma unroll
    for (int i = 0; i < 4; ++i) acc[1][i] = wmma_bf16(a1[1], b1[i], acc[1][i]);
    a1[0] = load_a_frag(Ar0 + kn1, Dn, lane);
    a1[1] = load_a_frag(Ar1 + kn1, Dn, lane);
#pragma unroll
    for (int i = 0; i < 4; ++i)
      b1[i] = load_b_frag(W + (size_t)(n0 + i * 16) * Dn + kn1, Dn, lane);
  }

  const int col = lane & 15;
  const int rbase = (lane >> 4) * 8;
#pragma unroll
  for (int mt = 0; mt < 2; ++mt) {
    const int m = m0 + mt * 16;
    const int b_ = m >> 11;
    const int t0 = m & (Tn - 1);
    if (mode < 2) {
      __bf16* O = (mode == 0) ? Q : K;
#pragma unroll
      for (int i = 0; i < 4; ++i) {
        const int dh = i * 16 + col;
#pragma unroll
        for (int r = 0; r < 8; ++r)
          O[((size_t)(b_ * Hn + h) * Tn + t0 + rbase + r) * DHn + dh] =
              (__bf16)acc[mt][i][r];
      }
    } else {  // V transposed: rows r contiguous along T -> one 16B store
#pragma unroll
      for (int i = 0; i < 4; ++i) {
        const int dh = i * 16 + col;
        v8bf pack;
#pragma unroll
        for (int r = 0; r < 8; ++r) pack[r] = (__bf16)acc[mt][i][r];
        *(v8bf*)(V + ((size_t)(b_ * Hn + h) * DHn + dh) * Tn + t0 + rbase) = pack;
      }
    }
  }
}

// ------------------------------------------------------- causal flash attention
// grid: (T/64, B*H), block 128 (4 waves); each wave owns a 16-query tile.
// K frags clustered -> partial loadcnt wait; V frags issued before softmax so
// the exp/shuffle VALU block hides their latency.
__global__ __launch_bounds__(128) void attn_kernel(const __bf16* __restrict__ Q,
                                                   const __bf16* __restrict__ Km,
                                                   const __bf16* __restrict__ Vt,
                                                   __bf16* __restrict__ Ctx) {
  __shared__ __bf16 ldsP[4][16 * 32];  // per-wave P tile (16 q x 32 k) bf16
  const int lane = threadIdx.x & 31;
  const int w    = threadIdx.x >> 5;
  const int bh   = blockIdx.y;
  const int q0   = (blockIdx.x * 4 + w) * 16;
  __bf16* Pt = ldsP[w];

  const int col   = lane & 15;
  const int rbase = (lane >> 4) * 8;

  const __bf16* Qb = Q + ((size_t)bh * Tn + q0) * DHn;
  const v16bf aq0 = load_a_frag(Qb, DHn, lane);       // dh 0..31
  const v16bf aq1 = load_a_frag(Qb + 32, DHn, lane);  // dh 32..63

  v8f acc0 = {}, acc1 = {}, acc2 = {}, acc3 = {};
  float mrun[8], srun[8];
#pragma unroll
  for (int r = 0; r < 8; ++r) { mrun[r] = -3.0e38f; srun[r] = 0.0f; }

  for (int kc = 0; kc < q0 + 16; kc += 32) {
    const __bf16* Kb = Km + ((size_t)bh * Tn + kc) * DHn;
    const __bf16* Vb = Vt + (size_t)bh * DHn * Tn + kc;
    if (kc + 32 < q0 + 16) {  // prefetch next K/V chunk -> global_prefetch_b8
      __builtin_prefetch(Kb + (size_t)32 * DHn, 0, 1);
      __builtin_prefetch(Vb + 32, 0, 1);
    }
    // cluster all K-fragment loads (distinct registers)
    const v16bf kb00 = load_b_frag(Kb, DHn, lane);
    const v16bf kb01 = load_b_frag(Kb + 32, DHn, lane);
    const v16bf kb10 = load_b_frag(Kb + (size_t)16 * DHn, DHn, lane);
    const v16bf kb11 = load_b_frag(Kb + (size_t)16 * DHn + 32, DHn, lane);
    // issue V-fragment loads now; consumed only after the softmax block
    const v16bf vb0 = load_b_frag(Vb + (size_t)0  * Tn, Tn, lane);
    const v16bf vb1 = load_b_frag(Vb + (size_t)16 * Tn, Tn, lane);
    const v16bf vb2 = load_b_frag(Vb + (size_t)32 * Tn, Tn, lane);
    const v16bf vb3 = load_b_frag(Vb + (size_t)48 * Tn, Tn, lane);

    // S = Q K^T : two 16x16 key tiles, K-dim = DH split 0..31 / 32..63
    v8f s0 = {}, s1 = {};
    s0 = wmma_bf16(aq0, kb00, s0);
    s0 = wmma_bf16(aq1, kb01, s0);
    s1 = wmma_bf16(aq0, kb10, s1);
    s1 = wmma_bf16(aq1, kb11, s1);

    const int key0 = kc + col, key1 = kc + 16 + col;
    float scale[8];
#pragma unroll
    for (int r = 0; r < 8; ++r) {
      const int qrow = q0 + rbase + r;
      float v0 = (key0 <= qrow) ? s0[r] * 0.125f : -3.0e38f;  // 1/sqrt(64), causal
      float v1 = (key1 <= qrow) ? s1[r] * 0.125f : -3.0e38f;
      float m = fmaxf(v0, v1);                 // row reduce within 16-lane half
      m = fmaxf(m, __shfl_xor(m, 1));
      m = fmaxf(m, __shfl_xor(m, 2));
      m = fmaxf(m, __shfl_xor(m, 4));
      m = fmaxf(m, __shfl_xor(m, 8));
      const float mnew = fmaxf(mrun[r], m);
      const float p0 = __expf(v0 - mnew);
      const float p1 = __expf(v1 - mnew);
      float ps = p0 + p1;
      ps += __shfl_xor(ps, 1);
      ps += __shfl_xor(ps, 2);
      ps += __shfl_xor(ps, 4);
      ps += __shfl_xor(ps, 8);
      const float sc = __expf(mrun[r] - mnew);
      srun[r] = srun[r] * sc + ps;
      mrun[r] = mnew;
      scale[r] = sc;
      s0[r] = p0;
      s1[r] = p1;
    }
#pragma unroll
    for (int r = 0; r < 8; ++r) {
      acc0[r] *= scale[r]; acc1[r] *= scale[r];
      acc2[r] *= scale[r]; acc3[r] *= scale[r];
    }
    // C-layout -> A-layout conversion through wave-private LDS slice
#pragma unroll
    for (int r = 0; r < 8; ++r) {
      const int row = rbase + r;
      Pt[row * 32 + col]      = (__bf16)s0[r];
      Pt[row * 32 + 16 + col] = (__bf16)s1[r];
    }
    asm volatile("s_wait_dscnt 0" ::: "memory");
    const v16bf pa = load_a_frag(Pt, 32, lane);  // P as 16x32 A fragment
    // ctx += P V  (V frags already resident)
    acc0 = wmma_bf16(pa, vb0, acc0);
    acc1 = wmma_bf16(pa, vb1, acc1);
    acc2 = wmma_bf16(pa, vb2, acc2);
    acc3 = wmma_bf16(pa, vb3, acc3);
  }

  // Store ctx head-major [B,H,T,DH]: flat-reinterprets as [B, T, D] for out-proj
  __bf16* Cb = Ctx + ((size_t)bh * Tn + q0) * DHn;
#pragma unroll
  for (int r = 0; r < 8; ++r) {
    const float inv = 1.0f / srun[r];
    const int t = rbase + r;
    Cb[t * DHn +  0 + col] = (__bf16)(acc0[r] * inv);
    Cb[t * DHn + 16 + col] = (__bf16)(acc1[r] * inv);
    Cb[t * DHn + 32 + col] = (__bf16)(acc2[r] * inv);
    Cb[t * DHn + 48 + col] = (__bf16)(acc3[r] * inv);
  }
}

// -------------------------------------------- out = ctx2 @ Wo.T + bo (f32 out)
// Same 32x64-per-wave ping-pong structure as qkv_proj.
__global__ __launch_bounds__(256) void out_proj_kernel(
    const __bf16* __restrict__ A, const __bf16* __restrict__ Wo,
    const float* __restrict__ bo, float* __restrict__ Out) {
  const int lane = threadIdx.x & 31;
  const int w    = threadIdx.x >> 5;
  const int m0   = blockIdx.x * 256 + w * 32;
  const int n0   = blockIdx.y * 64;
  const __bf16* Ar0 = A + (size_t)m0 * Dn;
  const __bf16* Ar1 = Ar0 + (size_t)16 * Dn;

  v8f acc[2][4] = {};
  v16bf a0[2], a1[2], b0[4], b1[4];
  a0[0] = load_a_frag(Ar0, Dn, lane);
  a0[1] = load_a_frag(Ar1, Dn, lane);
  a1[0] = load_a_frag(Ar0 + 32, Dn, lane);
  a1[1] = load_a_frag(Ar1 + 32, Dn, lane);
#pragma unroll
  for (int i = 0; i < 4; ++i) {
    b0[i] = load_b_frag(Wo + (size_t)(n0 + i * 16) * Dn, Dn, lane);
    b1[i] = load_b_frag(Wo + (size_t)(n0 + i * 16) * Dn + 32, Dn, lane);
  }

  for (int k = 0; k < Dn; k += 64) {
    const int kn0 = (k + 64 < Dn) ? k + 64 : 0;
    const int kn1 = (k + 96 < Dn) ? k + 96 : 0;
#pragma unroll
    for (int i = 0; i < 4; ++i) acc[0][i] = wmma_bf16(a0[0], b0[i], acc[0][i]);
#pragma unroll
    for (int i = 0; i < 4; ++i) acc[1][i] = wmma_bf16(a0[1], b0[i], acc[1][i]);
    a0[0] = load_a_frag(Ar0 + kn0, Dn, lane);
    a0[1] = load_a_frag(Ar1 + kn0, Dn, lane);
#pragma unroll
    for (int i = 0; i < 4; ++i)
      b0[i] = load_b_frag(Wo + (size_t)(n0 + i * 16) * Dn + kn0, Dn, lane);
#pragma unroll
    for (int i = 0; i < 4; ++i) acc[0][i] = wmma_bf16(a1[0], b1[i], acc[0][i]);
#pragma unroll
    for (int i = 0; i < 4; ++i) acc[1][i] = wmma_bf16(a1[1], b1[i], acc[1][i]);
    a1[0] = load_a_frag(Ar0 + kn1, Dn, lane);
    a1[1] = load_a_frag(Ar1 + kn1, Dn, lane);
#pragma unroll
    for (int i = 0; i < 4; ++i)
      b1[i] = load_b_frag(Wo + (size_t)(n0 + i * 16) * Dn + kn1, Dn, lane);
  }

  const int col = lane & 15;
  const int rbase = (lane >> 4) * 8;
#pragma unroll
  for (int mt = 0; mt < 2; ++mt) {
#pragma unroll
    for (int i = 0; i < 4; ++i) {
      const float bias = bo[n0 + i * 16 + col];
#pragma unroll
      for (int r = 0; r < 8; ++r)
        Out[(size_t)(m0 + mt * 16 + rbase + r) * Dn + n0 + i * 16 + col] =
            acc[mt][i][r] + bias;
    }
  }
}

extern "C" void kernel_launch(void* const* d_in, const int* in_sizes, int n_in,
                              void* d_out, int out_size, void* d_ws, size_t ws_size,
                              hipStream_t stream) {
  (void)in_sizes; (void)n_in; (void)out_size; (void)ws_size;
  const float* x  = (const float*)d_in[0];
  const float* Wq = (const float*)d_in[1];
  const float* Wk = (const float*)d_in[2];
  const float* Wv = (const float*)d_in[3];
  const float* Wo = (const float*)d_in[4];
  const float* bo = (const float*)d_in[5];

  // workspace carve (48 MiB total)
  char* p = (char*)d_ws;
  __bf16* Xb  = (__bf16*)p; p += (size_t)BTn * Dn * 2;  // x in bf16
  __bf16* Wqb = (__bf16*)p; p += (size_t)Dn * Dn * 2;
  __bf16* Wkb = (__bf16*)p; p += (size_t)Dn * Dn * 2;
  __bf16* Wvb = (__bf16*)p; p += (size_t)Dn * Dn * 2;
  __bf16* Wob = (__bf16*)p; p += (size_t)Dn * Dn * 2;
  __bf16* Qb  = (__bf16*)p; p += (size_t)BTn * Dn * 2;  // [B,H,T,DH]
  __bf16* Kb  = (__bf16*)p; p += (size_t)BTn * Dn * 2;  // [B,H,T,DH]
  __bf16* Vtb = (__bf16*)p; p += (size_t)BTn * Dn * 2;  // [B,H,DH,T]
  __bf16* Ctb = (__bf16*)p; p += (size_t)BTn * Dn * 2;  // ctx head-major

  const int nx = BTn * Dn, nw = Dn * Dn;
  cvt_f32_bf16<<<(nx / 4 + 255) / 256, 256, 0, stream>>>(x, Xb, nx);
  cvt_f32_bf16<<<(nw / 4 + 255) / 256, 256, 0, stream>>>(Wq, Wqb, nw);
  cvt_f32_bf16<<<(nw / 4 + 255) / 256, 256, 0, stream>>>(Wk, Wkb, nw);
  cvt_f32_bf16<<<(nw / 4 + 255) / 256, 256, 0, stream>>>(Wv, Wvb, nw);
  cvt_f32_bf16<<<(nw / 4 + 255) / 256, 256, 0, stream>>>(Wo, Wob, nw);

  qkv_proj_kernel<<<dim3(BTn / 256, Hn, 3), 256, 0, stream>>>(Xb, Wqb, Wkb, Wvb,
                                                              Qb, Kb, Vtb);
  attn_kernel<<<dim3(Tn / 64, Bn * Hn), 128, 0, stream>>>(Qb, Kb, Vtb, Ctb);
  out_proj_kernel<<<dim3(BTn / 256, Dn / 64), 256, 0, stream>>>(Ctb, Wob, bo,
                                                                (float*)d_out);
}
